// ContactGNN_22136261444097
// MI455X (gfx1250) — compile-verified, matrix operand order checked by reading, not run on hardware
//
#include <hip/hip_runtime.h>
#include <hip/hip_bf16.h>

#define Mnodes 4096
#define Eedges 262144
#define HC 256
#define ENC 128
#define LAT 128

typedef __attribute__((ext_vector_type(16))) __bf16 v16bf;
typedef __attribute__((ext_vector_type(8)))  __bf16 v8bf;
typedef __attribute__((ext_vector_type(8)))  float  v8f;

// ---------- helpers ----------
__device__ __forceinline__ __bf16 f2bf(float f) {
  unsigned u = __builtin_bit_cast(unsigned, f);
  unsigned r = u + 0x7FFFu + ((u >> 16) & 1u);        // round-to-nearest-even
  unsigned short h = (unsigned short)(r >> 16);
  return __builtin_bit_cast(__bf16, h);
}
// monotone float<->int mapping so atomicMax(int) == float max
__device__ __forceinline__ int ford(float f) {
  int i = __builtin_bit_cast(int, f);
  return i ^ ((i >> 31) & 0x7fffffff);
}
__device__ __forceinline__ float ford_inv(int i) {
  int j = i ^ ((i >> 31) & 0x7fffffff);
  return __builtin_bit_cast(float, j);
}
#define FORD_NEG_INF ((int)0x807FFFFF)   // ford(-inf)

__device__ __forceinline__ v16bf glue(v8bf lo, v8bf hi) {
  return __builtin_shufflevector(lo, hi, 0,1,2,3,4,5,6,7,8,9,10,11,12,13,14,15);
}

// ---------- node encoder: h = relu(x @ W + b), K=10, bf16 output ----------
__global__ void encoder_kernel(const float* __restrict__ x, const float* __restrict__ W,
                               const float* __restrict__ b, __bf16* __restrict__ h) {
  int idx = blockIdx.x * blockDim.x + threadIdx.x;         // 4096*128
  if (idx >= Mnodes * ENC) return;
  int m = idx >> 7, n = idx & 127;
  float acc = b[n];
  #pragma unroll
  for (int k = 0; k < 10; ++k) acc += x[m * 10 + k] * W[k * ENC + n];
  h[idx] = f2bf(acc > 0.f ? acc : 0.f);
}

// ---------- weight prep: W[K,N] f32 -> Wt[N,K] bf16 ----------
__global__ void wprep_kernel(const float* __restrict__ W, __bf16* __restrict__ Wt,
                             int K, int N) {
  int idx = blockIdx.x * blockDim.x + threadIdx.x;
  if (idx >= K * N) return;
  int k = idx / N, n = idx % N;
  Wt[n * K + k] = f2bf(W[idx]);
}

// ---------- activation convert: f32 -> bf16 ----------
__global__ void cvt_kernel(const float* __restrict__ in, __bf16* __restrict__ out, int cnt) {
  int idx = blockIdx.x * blockDim.x + threadIdx.x;
  if (idx < cnt) out[idx] = f2bf(in[idx]);
}

// ---------- bf16 WMMA GEMM, both operands K-contiguous ----------
// A [M,K] bf16 row-major; Bt [N,K] bf16 row-major (i.e. B transposed).
// C = act(A @ Bt^T + bias). Each wave computes a 32x32 tile via 2x2 WMMA tiles.
// Cf (f32) and Cbf (bf16) outputs each optional (null to skip).
__global__ void gemm_bf16_tt(const __bf16* __restrict__ A, const __bf16* __restrict__ Bt,
                             float* __restrict__ Cf, __bf16* __restrict__ Cbf,
                             const float* __restrict__ bias,
                             int M, int N, int K, int do_relu) {
  const int lane = threadIdx.x & 31;
  const int wave = threadIdx.x >> 5;
  const int tn2 = blockIdx.x * 4 + wave;   // 32-wide N tile per wave
  const int tm2 = blockIdx.y;              // 32-tall M tile
  const int l = lane & 15, hi = lane >> 4;
  const __bf16* pa0 = A  + (size_t)(tm2 * 32 + l) * K;
  const __bf16* pa1 = pa0 + (size_t)16 * K;
  const __bf16* pb0 = Bt + (size_t)(tn2 * 32 + l) * K;
  const __bf16* pb1 = pb0 + (size_t)16 * K;
  v8f acc00 = {}, acc01 = {}, acc10 = {}, acc11 = {};
  for (int k0 = 0; k0 < K; k0 += 32) {
    const int ka = k0 + hi * 8;            // fragment: K = ka+{0..7}, ka+16+{0..7}
    v16bf a0 = glue(*(const v8bf*)(pa0 + ka), *(const v8bf*)(pa0 + ka + 16));
    v16bf a1 = glue(*(const v8bf*)(pa1 + ka), *(const v8bf*)(pa1 + ka + 16));
    v16bf b0 = glue(*(const v8bf*)(pb0 + ka), *(const v8bf*)(pb0 + ka + 16));
    v16bf b1 = glue(*(const v8bf*)(pb1 + ka), *(const v8bf*)(pb1 + ka + 16));
    acc00 = __builtin_amdgcn_wmma_f32_16x16x32_bf16(false, a0, false, b0, (short)0, acc00, false, false);
    acc01 = __builtin_amdgcn_wmma_f32_16x16x32_bf16(false, a0, false, b1, (short)0, acc01, false, false);
    acc10 = __builtin_amdgcn_wmma_f32_16x16x32_bf16(false, a1, false, b0, (short)0, acc10, false, false);
    acc11 = __builtin_amdgcn_wmma_f32_16x16x32_bf16(false, a1, false, b1, (short)0, acc11, false, false);
  }
  // C/D layout: VGPR r -> (M = hi*8 + r, N = l) within each 16x16 tile
  const int n0 = tn2 * 32 + l, n1 = n0 + 16;
  const float bv0 = bias ? bias[n0] : 0.f;
  const float bv1 = bias ? bias[n1] : 0.f;
  #pragma unroll
  for (int r = 0; r < 8; ++r) {
    const int m0 = tm2 * 32 + hi * 8 + r, m1 = m0 + 16;
    float v00 = acc00[r] + bv0, v01 = acc01[r] + bv1;
    float v10 = acc10[r] + bv0, v11 = acc11[r] + bv1;
    if (do_relu) {
      v00 = v00 > 0.f ? v00 : 0.f; v01 = v01 > 0.f ? v01 : 0.f;
      v10 = v10 > 0.f ? v10 : 0.f; v11 = v11 > 0.f ? v11 : 0.f;
    }
    if (Cf) {
      Cf[(size_t)m0 * N + n0] = v00; Cf[(size_t)m0 * N + n1] = v01;
      Cf[(size_t)m1 * N + n0] = v10; Cf[(size_t)m1 * N + n1] = v11;
    }
    if (Cbf) {
      Cbf[(size_t)m0 * N + n0] = f2bf(v00); Cbf[(size_t)m0 * N + n1] = f2bf(v01);
      Cbf[(size_t)m1 * N + n0] = f2bf(v10); Cbf[(size_t)m1 * N + n1] = f2bf(v11);
    }
  }
}

// ---------- GAT init: out = bias (scatter accumulates on top), lmax=-inf, denom=0 ----------
__global__ void gat_init_kernel(float* __restrict__ out, const float* __restrict__ bias,
                                int* __restrict__ lmax, float* __restrict__ denom) {
  int idx = blockIdx.x * blockDim.x + threadIdx.x;
  if (idx < Mnodes * HC) out[idx] = bias[idx & (HC - 1)];
  if (idx < Mnodes * 4) { lmax[idx] = FORD_NEG_INF; denom[idx] = 0.f; }
}

// ---------- edge pass 1: logits + segment max (one wave per edge) ----------
__global__ void edge_logits_kernel(const int* __restrict__ ei, const float* __restrict__ eattr,
                                   const float* __restrict__ xl, const float* __restrict__ xr,
                                   const float* __restrict__ We, const float* __restrict__ att,
                                   float* __restrict__ logits, int* __restrict__ lmax) {
  const int lane = threadIdx.x & 31;
  const int wave = threadIdx.x >> 5;
  const int e = blockIdx.x * 8 + wave;
  if (e >= Eedges) return;
  const int s = ei[e], d = ei[Eedges + e];
  const float ea = eattr[e];
  const int c0 = lane * 8;                     // 8 channels/lane; lanes 8h..8h+7 = head h
  const float* pl = xl + (size_t)s * HC + c0;
  const float* pr = xr + (size_t)d * HC + c0;
  float part = 0.f;
  #pragma unroll
  for (int j = 0; j < 8; ++j) {
    float v = pl[j] + pr[j] + ea * We[c0 + j];
    v = v > 0.f ? v : 0.2f * v;                // leaky_relu
    part += v * att[c0 + j];                   // att flat [H*C], index == channel
  }
  #pragma unroll
  for (int o = 1; o < 8; o <<= 1) part += __shfl_xor(part, o, 32);
  if ((lane & 7) == 0) {
    const int h = lane >> 3;
    logits[(size_t)e * 4 + h] = part;
    atomicMax(&lmax[d * 4 + h], ford(part));
  }
}

// ---------- edge pass 2: exp + segment sum (one thread per edge-head) ----------
__global__ void edge_softmax_kernel(const int* __restrict__ ei, const float* __restrict__ logits,
                                    const int* __restrict__ lmax, float* __restrict__ exbuf,
                                    float* __restrict__ denom) {
  int idx = blockIdx.x * blockDim.x + threadIdx.x;
  if (idx >= Eedges * 4) return;
  int e = idx >> 2, h = idx & 3;
  int d = ei[Eedges + e];
  float v = __expf(logits[idx] - ford_inv(lmax[d * 4 + h]));
  exbuf[idx] = v;
  atomicAdd(&denom[d * 4 + h], v);
}

// ---------- edge pass 3: weighted scatter-add of messages (one wave per edge) ----------
__global__ void edge_message_kernel(const int* __restrict__ ei, const float* __restrict__ xl,
                                    const float* __restrict__ exbuf, const float* __restrict__ denom,
                                    float* __restrict__ out) {
  const int lane = threadIdx.x & 31;
  const int wave = threadIdx.x >> 5;
  const int e = blockIdx.x * 8 + wave;
  if (e >= Eedges) return;
  const int s = ei[e], d = ei[Eedges + e];
  const int c0 = lane * 8, h = lane >> 3;
  const float alpha = exbuf[(size_t)e * 4 + h] / (denom[d * 4 + h] + 1e-16f);
  const float* pl = xl + (size_t)s * HC + c0;
  float* po = out + (size_t)d * HC + c0;
  #pragma unroll
  for (int j = 0; j < 8; ++j) atomicAdd(&po[j], pl[j] * alpha);
}

// ---------- host-side orchestration ----------
extern "C" void kernel_launch(void* const* d_in, const int* in_sizes, int n_in,
                              void* d_out, int out_size, void* d_ws, size_t ws_size,
                              hipStream_t stream) {
  (void)in_sizes; (void)n_in; (void)out_size; (void)ws_size;
  const float* x     = (const float*)d_in[0];
  const int*   ei    = (const int*)  d_in[1];
  const float* eattr = (const float*)d_in[2];
  const float* enc_W = (const float*)d_in[3];
  const float* enc_b = (const float*)d_in[4];
  const float* g1_Wl = (const float*)d_in[5];
  const float* g1_Wr = (const float*)d_in[6];
  const float* g1_We = (const float*)d_in[7];
  const float* g1_at = (const float*)d_in[8];
  const float* g1_b  = (const float*)d_in[9];
  const float* u1_W1 = (const float*)d_in[10];
  const float* u1_b1 = (const float*)d_in[11];
  const float* u1_W2 = (const float*)d_in[12];
  const float* u1_b2 = (const float*)d_in[13];
  const float* g2_Wl = (const float*)d_in[14];
  const float* g2_Wr = (const float*)d_in[15];
  const float* g2_We = (const float*)d_in[16];
  const float* g2_at = (const float*)d_in[17];
  const float* g2_b  = (const float*)d_in[18];
  const float* u2_W1 = (const float*)d_in[19];
  const float* u2_b1 = (const float*)d_in[20];
  const float* u2_W2 = (const float*)d_in[21];
  const float* u2_b2 = (const float*)d_in[22];
  float* out = (float*)d_out;

  // ---- workspace layout (~24 MB, L2-resident) ----
  const int MHC = Mnodes * HC, E4 = Eedges * 4, MH = Mnodes * 4;
  float* ws = (float*)d_ws;
  float* xl    = ws;                    // [4096,256] f32
  float* xr    = xl + MHC;
  float* gout  = xr + MHC;
  float* logit = gout + MHC;            // [E,4]
  float* exbuf = logit + E4;
  float* denom = exbuf + E4;            // [4096,4]
  int*   lmax  = (int*)(denom + MH);    // [4096,4]
  __bf16* hA    = (__bf16*)(lmax + MH); // [4096,128] bf16
  __bf16* hB    = hA + Mnodes * ENC;
  __bf16* goutb = hB + Mnodes * ENC;    // [4096,256] bf16
  __bf16* midb  = goutb + MHC;          // [4096,256] bf16
  __bf16* w1lt  = midb + MHC;           // g1_Wl^T [256,128]
  __bf16* w1rt  = w1lt + HC * ENC;
  __bf16* wu11t = w1rt + HC * ENC;      // u1_W1^T [256,256]
  __bf16* wu12t = wu11t + 256 * HC;     // u1_W2^T [128,256]
  __bf16* w2lt  = wu12t + LAT * 256;    // g2_Wl^T [256,128]
  __bf16* w2rt  = w2lt + HC * LAT;
  __bf16* wu21t = w2rt + HC * LAT;      // u2_W1^T [256,256]
  __bf16* wu22t = wu21t + 256 * HC;     // u2_W2^T [128,256]

  const dim3 blk128(128), blk256(256);
  const dim3 gN256(HC / 128, Mnodes / 32);      // N=256: 2 x 128 blocks
  const dim3 gN128(LAT / 128, Mnodes / 32);     // N=128: 1 x 128
  const dim3 gOut(Mnodes / 128, Mnodes / 32);   // N=4096: 32 x 128
  const int edgeBlocks = Eedges / 8;

  // ---- one-shot prep: encoder + weight transposes (all cheap) ----
  encoder_kernel<<<(Mnodes * ENC + 255) / 256, blk256, 0, stream>>>(x, enc_W, enc_b, hA);
  wprep_kernel<<<(ENC * HC + 255) / 256, blk256, 0, stream>>>(g1_Wl, w1lt, ENC, HC);
  wprep_kernel<<<(ENC * HC + 255) / 256, blk256, 0, stream>>>(g1_Wr, w1rt, ENC, HC);
  wprep_kernel<<<(HC * 256 + 255) / 256, blk256, 0, stream>>>(u1_W1, wu11t, HC, 256);
  wprep_kernel<<<(256 * LAT + 255) / 256, blk256, 0, stream>>>(u1_W2, wu12t, 256, LAT);
  wprep_kernel<<<(LAT * HC + 255) / 256, blk256, 0, stream>>>(g2_Wl, w2lt, LAT, HC);
  wprep_kernel<<<(LAT * HC + 255) / 256, blk256, 0, stream>>>(g2_Wr, w2rt, LAT, HC);
  wprep_kernel<<<(HC * 256 + 255) / 256, blk256, 0, stream>>>(u2_W1, wu21t, HC, 256);
  wprep_kernel<<<(256 * LAT + 255) / 256, blk256, 0, stream>>>(u2_W2, wu22t, 256, LAT);

  // ---- GAT layer 1 ----
  gemm_bf16_tt<<<gN256, blk128, 0, stream>>>(hA, w1lt, xl, nullptr, nullptr, Mnodes, HC, ENC, 0);
  gemm_bf16_tt<<<gN256, blk128, 0, stream>>>(hA, w1rt, xr, nullptr, nullptr, Mnodes, HC, ENC, 0);
  gat_init_kernel<<<(MHC + 255) / 256, blk256, 0, stream>>>(gout, g1_b, lmax, denom);
  edge_logits_kernel<<<edgeBlocks, blk256, 0, stream>>>(ei, eattr, xl, xr, g1_We, g1_at, logit, lmax);
  edge_softmax_kernel<<<(E4 + 255) / 256, blk256, 0, stream>>>(ei, logit, lmax, exbuf, denom);
  edge_message_kernel<<<edgeBlocks, blk256, 0, stream>>>(ei, xl, exbuf, denom, gout);
  cvt_kernel<<<(MHC + 255) / 256, blk256, 0, stream>>>(gout, goutb, MHC);
  // update MLP 1 (bf16 chained outputs)
  gemm_bf16_tt<<<gN256, blk128, 0, stream>>>(goutb, wu11t, nullptr, midb, u1_b1, Mnodes, 256, HC, 1);
  gemm_bf16_tt<<<gN128, blk128, 0, stream>>>(midb, wu12t, nullptr, hB, u1_b2, Mnodes, LAT, 256, 1);

  // ---- GAT layer 2 ----
  gemm_bf16_tt<<<gN256, blk128, 0, stream>>>(hB, w2lt, xl, nullptr, nullptr, Mnodes, HC, LAT, 0);
  gemm_bf16_tt<<<gN256, blk128, 0, stream>>>(hB, w2rt, xr, nullptr, nullptr, Mnodes, HC, LAT, 0);
  gat_init_kernel<<<(MHC + 255) / 256, blk256, 0, stream>>>(gout, g2_b, lmax, denom);
  edge_logits_kernel<<<edgeBlocks, blk256, 0, stream>>>(ei, eattr, xl, xr, g2_We, g2_at, logit, lmax);
  edge_softmax_kernel<<<(E4 + 255) / 256, blk256, 0, stream>>>(ei, logit, lmax, exbuf, denom);
  edge_message_kernel<<<edgeBlocks, blk256, 0, stream>>>(ei, xl, exbuf, denom, gout);
  cvt_kernel<<<(MHC + 255) / 256, blk256, 0, stream>>>(gout, goutb, MHC);
  // update MLP 2 -> final node features (bf16, reuse hA)
  gemm_bf16_tt<<<gN256, blk128, 0, stream>>>(goutb, wu21t, nullptr, midb, u2_b1, Mnodes, 256, HC, 1);
  gemm_bf16_tt<<<gN128, blk128, 0, stream>>>(midb, wu22t, nullptr, hA, u2_b2, Mnodes, LAT, 256, 1);

  // ---- contact map: out = h @ h^T (A and Bt are the same bf16 array) ----
  gemm_bf16_tt<<<gOut, blk128, 0, stream>>>(hA, hA, out, nullptr, nullptr, Mnodes, Mnodes, LAT, 0);
}
